// GCN_76012331205028
// MI455X (gfx1250) — matrix operand order repaired and checked by reading, not compiled, not run on hardware
//
#include <hip/hip_runtime.h>

#define N_NODES 100000
#define N_EDGES 3200000

typedef __attribute__((ext_vector_type(16))) _Float16 v16h;
typedef __attribute__((ext_vector_type(8)))  _Float16 v8h;
typedef __attribute__((ext_vector_type(8)))  float    v8f;
typedef __attribute__((ext_vector_type(4)))  int      v4i;

#if defined(__has_builtin)
#if __has_builtin(__builtin_amdgcn_global_load_async_to_lds_b128)
#define GCN_HAVE_ASYNC_LDS 1
#endif
#endif

__device__ __forceinline__ void gcn_wait_async0() {
#if defined(__has_builtin) && __has_builtin(__builtin_amdgcn_s_wait_asynccnt)
    __builtin_amdgcn_s_wait_asynccnt(0);
#else
    asm volatile("s_wait_asynccnt 0x0" ::: "memory");
#endif
}

// ---------------------------------------------------------------- degrees ---
__global__ void gcn_deg_kernel(const int* __restrict__ src, const int* __restrict__ dst,
                               float* __restrict__ deg_out, float* __restrict__ deg_in, int E) {
    int e = blockIdx.x * blockDim.x + threadIdx.x;
    if (e < E) {
        atomicAdd(&deg_out[src[e]], 1.0f);
        atomicAdd(&deg_in[dst[e]], 1.0f);
    }
}

__global__ void gcn_invsqrt_kernel(const float* __restrict__ dego, float* __restrict__ invo,
                                   const float* __restrict__ degi, float* __restrict__ invi, int n) {
    int i = blockIdx.x * blockDim.x + threadIdx.x;
    if (i < n) {
        float a = dego[i]; invo[i] = a > 0.0f ? rsqrtf(a) : 0.0f;
        float b = degi[i]; invi[i] = b > 0.0f ? rsqrtf(b) : 0.0f;
    }
}

// ------------------------------------------------- edge scatter (128 chan) ---
// One wave per edge; lane l handles channels [4l, 4l+4). Gathers hit L2
// (feature matrix is L2-resident), accumulation via global_atomic_add_f32.
__global__ void gcn_scatter128_kernel(const float* __restrict__ feat, const int* __restrict__ src,
                                      const int* __restrict__ dst, const float* __restrict__ invo,
                                      const float* __restrict__ invi, float* __restrict__ agg, int E) {
    int e = blockIdx.x * 8 + (threadIdx.x >> 5);
    int lane = threadIdx.x & 31;
    if (e >= E) return;
    int s = src[e], d = dst[e];
    float w = invo[s] * invi[d];
    const float4 v = ((const float4*)(feat + (size_t)s * 128))[lane];
    float* o = agg + (size_t)d * 128 + lane * 4;
    atomicAdd(o + 0, v.x * w);
    atomicAdd(o + 1, v.y * w);
    atomicAdd(o + 2, v.z * w);
    atomicAdd(o + 3, v.w * w);
}

// ---------------------------------------------------------- f32 -> f16 cvt ---
__global__ void gcn_cvt_f16_kernel(const float* __restrict__ in, _Float16* __restrict__ out,
                                   size_t n8) {
    size_t i = (size_t)blockIdx.x * blockDim.x + threadIdx.x;
    if (i >= n8) return;
    const float4* p = (const float4*)in + i * 2;
    float4 a = p[0], b = p[1];
    v8h o = { (_Float16)a.x, (_Float16)a.y, (_Float16)a.z, (_Float16)a.w,
              (_Float16)b.x, (_Float16)b.y, (_Float16)b.z, (_Float16)b.w };
    *(v8h*)(out + i * 8) = o;
}

// W [K,N] f32 (row-major, as in x @ W) -> Wt [N,K] f16 so a WMMA B-column is
// a contiguous run of K halves.
__global__ void gcn_transpose_cvt_kernel(const float* __restrict__ W, _Float16* __restrict__ Wt,
                                         int K, int N) {
    int n = blockIdx.x * blockDim.x + threadIdx.x;
    int k = blockIdx.y;
    if (n < N) Wt[(size_t)n * K + k] = (_Float16)W[(size_t)k * N + n];
}

// --------------------------------------------- post-scatter bias+relu dual ---
__global__ void gcn_bias_relu_dual_kernel(const float* __restrict__ agg, const float* __restrict__ bias,
                                          float* __restrict__ of, _Float16* __restrict__ oh, int total) {
    int i = blockIdx.x * blockDim.x + threadIdx.x;
    if (i >= total) return;
    float v = agg[i] + bias[i & 127];
    v = fmaxf(v, 0.0f);
    of[i] = v;                 // fp32 copy (h_last output)
    oh[i] = (_Float16)v;       // f16 copy feeding the MLP WMMA GEMM
}

// ----------------------------------------------------------- WMMA GEMM ------
// C[M,N] = relu?(A[M,K] * Bt[N,K]^T + bias). One block = one 16-row M tile;
// A tile staged into LDS with GLOBAL_LOAD_ASYNC_TO_LDS (ASYNCcnt) when the
// toolchain exposes it; 4 waves sweep 16-col N tiles. B fragments for a tile
// are preloaded as a register array so the WMMA chain issues back-to-back.
// Fragment layouts used here, contiguous k in A and B (16-bit A 16x32; B = column per lane,
// K split at 16 between lane halves).
template <int K, int NDIM, bool RELU, bool OUT_F16>
__global__ void __launch_bounds__(128)
gcn_wmma_gemm_kernel(const _Float16* __restrict__ A, const _Float16* __restrict__ Bt,
                     const float* __restrict__ bias, float* __restrict__ Cf,
                     _Float16* __restrict__ Ch) {
    constexpr int LDA = K + 8;                 // pad to avoid LDS bank conflicts
    __shared__ _Float16 As[16 * LDA];
    const int m0   = blockIdx.x * 16;
    const int wave = threadIdx.x >> 5;
    const int lane = threadIdx.x & 31;
    const int row  = lane & 15;
    const int hi   = lane >> 4;

    // stage the 16 x K tile (rows contiguous in global A) into LDS
    const _Float16* Atile = A + (size_t)m0 * K;
#if GCN_HAVE_ASYNC_LDS
    for (int idx = threadIdx.x; idx < 16 * (K / 8); idx += 128) {
        int r = idx / (K / 8);
        int c = (idx % (K / 8)) * 8;
        __builtin_amdgcn_global_load_async_to_lds_b128(
            (__attribute__((address_space(1))) v4i*)(Atile + r * K + c),
            (__attribute__((address_space(3))) v4i*)(As + r * LDA + c),
            0, 0);
    }
    gcn_wait_async0();
    __syncthreads();
#else
    for (int idx = threadIdx.x; idx < 16 * (K / 8); idx += 128) {
        int r = idx / (K / 8);
        int c = (idx % (K / 8)) * 8;
        *(v8h*)(As + r * LDA + c) = *(const v8h*)(Atile + r * K + c);
    }
    __syncthreads();
#endif

    // per-lane A fragments, reused across all N tiles
    v16h afrag[K / 32];
    const _Float16* asrc = As + row * LDA + hi * 16;
#pragma unroll
    for (int kk = 0; kk < K / 32; ++kk) {
        ((v8h*)&afrag[kk])[0] = *(const v8h*)(asrc + kk * 32);       // K = 0..7   (+16*hi)
        ((v8h*)&afrag[kk])[1] = *(const v8h*)(asrc + kk * 32 + 8);   // K = 8..15  (+16*hi)
    }

    for (int nt = wave; nt * 16 < NDIM; nt += 4) {
        const int n = nt * 16 + row;
        const _Float16* brow = Bt + (size_t)n * K + hi * 16;

        // preload the full B panel for this tile so the WMMA chain is not
        // interleaved with load->wait pairs
        v16h bfrag[K / 32];
#pragma unroll
        for (int kk = 0; kk < K / 32; ++kk)
            bfrag[kk] = *(const v16h*)(brow + kk * 32);

        const float bn = bias ? bias[n] : 0.0f;
        v8f acc = {};
#pragma unroll
        for (int kk = 0; kk < K / 32; ++kk) {
            acc = __builtin_amdgcn_wmma_f32_16x16x32_f16(
                false, afrag[kk], false, bfrag[kk], (short)0, acc, false, false);
        }
#pragma unroll
        for (int r = 0; r < 8; ++r) {
            int m = m0 + r + 8 * hi;           // C/D layout: VGPR r -> M=r (+8 for hi lanes)
            float v = acc[r] + bn;
            if (RELU) v = fmaxf(v, 0.0f);
            if (OUT_F16) Ch[(size_t)m * NDIM + n] = (_Float16)v;
            else         Cf[(size_t)m * NDIM + n] = v;
        }
    }
}

// ---------------------------------------------------------------- launch ----
extern "C" void kernel_launch(void* const* d_in, const int* in_sizes, int n_in,
                              void* d_out, int out_size, void* d_ws, size_t ws_size,
                              hipStream_t stream) {
    (void)in_sizes; (void)n_in; (void)out_size; (void)ws_size;
    const float* x    = (const float*)d_in[0];
    const int*   ei   = (const int*)d_in[1];
    const float* W0   = (const float*)d_in[2];
    const float* b0   = (const float*)d_in[3];
    const float* W1   = (const float*)d_in[4];
    const float* b1   = (const float*)d_in[5];
    const float* fc1w = (const float*)d_in[6];
    const float* fc1b = (const float*)d_in[7];
    const float* fc2w = (const float*)d_in[8];
    const float* fc2b = (const float*)d_in[9];
    const int* src = ei;
    const int* dst = ei + N_EDGES;

    char* ws = (char*)d_ws;
    size_t off = 0;
    auto alloc = [&](size_t bytes) -> void* {
        void* p = ws + off;
        off = (off + bytes + 255) & ~(size_t)255;
        return p;
    };
    float*    deg_out = (float*)alloc((size_t)N_NODES * 4);
    float*    deg_in  = (float*)alloc((size_t)N_NODES * 4);
    float*    inv_out = (float*)alloc((size_t)N_NODES * 4);
    float*    inv_in  = (float*)alloc((size_t)N_NODES * 4);
    float*    bufA    = (float*)alloc((size_t)N_NODES * 128 * 4);    // agg0, later t
    _Float16* bufHa   = (_Float16*)alloc((size_t)N_NODES * 128 * 2); // agg0 f16, later h_last f16
    void*     bufHb   = alloc((size_t)N_NODES * 256 * 2);            // h1 f16, later agg1 f32
    _Float16* m1      = (_Float16*)alloc((size_t)N_NODES * 64 * 2);
    _Float16* W0t     = (_Float16*)alloc(256 * 128 * 2);
    _Float16* W1t     = (_Float16*)alloc(128 * 256 * 2);
    _Float16* fc1t    = (_Float16*)alloc(64 * 128 * 2);
    _Float16* fc2t    = (_Float16*)alloc(64 * 64 * 2);

    float* out_f32    = (float*)d_out;                         // [100000, 64]
    float* h_last_f32 = (float*)d_out + (size_t)N_NODES * 64;  // [100000, 128]

    // --- edge normalization ---
    (void)hipMemsetAsync(deg_out, 0, (size_t)N_NODES * 4, stream);
    (void)hipMemsetAsync(deg_in,  0, (size_t)N_NODES * 4, stream);
    gcn_deg_kernel<<<(N_EDGES + 255) / 256, 256, 0, stream>>>(src, dst, deg_out, deg_in, N_EDGES);
    gcn_invsqrt_kernel<<<(N_NODES + 255) / 256, 256, 0, stream>>>(deg_out, inv_out, deg_in, inv_in, N_NODES);

    // --- weights -> transposed f16 ---
    gcn_transpose_cvt_kernel<<<dim3(1, 128), 256, 0, stream>>>(W0,   W0t, 128, 256);
    gcn_transpose_cvt_kernel<<<dim3(1, 256), 256, 0, stream>>>(W1,   W1t, 256, 128);
    gcn_transpose_cvt_kernel<<<dim3(1, 128), 256, 0, stream>>>(fc1w, fc1t, 128, 64);
    gcn_transpose_cvt_kernel<<<dim3(1, 64),  256, 0, stream>>>(fc2w, fc2t, 64, 64);

    // --- layer 0: agg0 = A_hat * x ; h1 = relu(agg0 @ W0 + b0) (f16 out) ---
    (void)hipMemsetAsync(bufA, 0, (size_t)N_NODES * 128 * 4, stream);
    gcn_scatter128_kernel<<<N_EDGES / 8, 256, 0, stream>>>(x, src, dst, inv_out, inv_in, bufA, N_EDGES);
    gcn_cvt_f16_kernel<<<((N_NODES * 128 / 8) + 255) / 256, 256, 0, stream>>>(
        bufA, bufHa, (size_t)N_NODES * 128 / 8);
    gcn_wmma_gemm_kernel<128, 256, true, true><<<N_NODES / 16, 128, 0, stream>>>(
        bufHa, W0t, b0, nullptr, (_Float16*)bufHb);

    // --- layer 1: t = h1 @ W1 (f32) ; agg1 = A_hat * t ; h_last = relu(agg1 + b1) ---
    gcn_wmma_gemm_kernel<256, 128, false, false><<<N_NODES / 16, 128, 0, stream>>>(
        (_Float16*)bufHb, W1t, nullptr, bufA, nullptr);
    (void)hipMemsetAsync(bufHb, 0, (size_t)N_NODES * 128 * 4, stream);   // reuse as fp32 agg1
    gcn_scatter128_kernel<<<N_EDGES / 8, 256, 0, stream>>>(bufA, src, dst, inv_out, inv_in,
                                                           (float*)bufHb, N_EDGES);
    gcn_bias_relu_dual_kernel<<<(N_NODES * 128 + 255) / 256, 256, 0, stream>>>(
        (const float*)bufHb, b1, h_last_f32, bufHa, N_NODES * 128);

    // --- MLP: out = relu(h_last @ fc1 + b) @ fc2 + b ---
    gcn_wmma_gemm_kernel<128, 64, true, true><<<N_NODES / 16, 128, 0, stream>>>(
        bufHa, fc1t, fc1b, nullptr, m1);
    gcn_wmma_gemm_kernel<64, 64, false, false><<<N_NODES / 16, 128, 0, stream>>>(
        m1, fc2t, fc2b, out_f32, nullptr);
}
